// MultiHeadAttention_652835029015
// MI455X (gfx1250) — matrix operand order repaired and checked by reading, not compiled, not run on hardware
//
#include <hip/hip_runtime.h>
#include <hip/hip_bf16.h>

// ---------------------------------------------------------------------------
// MHA for gfx1250 (MI455X): bf16 WMMA everywhere, fp32 accumulate.
// B=16, N=1024, E=768, H=12, D=64
// Workgroup-shared B-operand tiles in LDS staged with
// GLOBAL_LOAD_ASYNC_TO_LDS_B128 (ASYNCcnt), double-buffered.
// GEMM waves compute 32x64 (2 A-frags x 4 B-frags -> 8 WMMAs/k-step) so each
// LDS B-fragment read feeds two WMMAs (halves LDS bw + wait frequency).
// ---------------------------------------------------------------------------

typedef __attribute__((ext_vector_type(16))) __bf16 v16bf;
typedef __attribute__((ext_vector_type(8)))  float  v8f;
typedef __attribute__((ext_vector_type(4)))  int    v4i;
typedef unsigned short ushort_t;

#define Bb 16
#define Nn 1024
#define Ee 768
#define Hh 12
#define Dd 64
#define Mrows (Bb * Nn)       // 16384
#define QKVN  (3 * Ee)        // 2304

struct alignas(16) US8 { unsigned short s[8]; };

#if __has_builtin(__builtin_amdgcn_global_load_async_to_lds_b128)
#define HAS_ASYNC 1
#else
#define HAS_ASYNC 0
#endif

// 16B memory -> LDS copy. Async (direct-to-LDS, ASYNCcnt) when available.
__device__ inline void cp16_to_lds(const void* g, void* lds_generic) {
#if HAS_ASYNC
    __builtin_amdgcn_global_load_async_to_lds_b128(
        (__attribute__((address_space(1))) v4i*)(uintptr_t)g,
        (__attribute__((address_space(3))) v4i*)(unsigned)(uintptr_t)lds_generic,
        0, 0);
#else
    *(US8*)lds_generic = *(const US8*)g;
#endif
}

template <int N> __device__ inline void wait_async() {
#if HAS_ASYNC
#if __has_builtin(__builtin_amdgcn_s_wait_asynccnt)
    __builtin_amdgcn_s_wait_asynccnt(N);
#else
    asm volatile("s_wait_asynccnt %0" ::"i"(N) : "memory");
#endif
#endif
}

__device__ inline unsigned short f2bf(float f) {
    unsigned int u = __builtin_bit_cast(unsigned int, f);
    unsigned int r = u + 0x7fffu + ((u >> 16) & 1u);
    return (unsigned short)(r >> 16);
}

// Load 8 contiguous bf16 (16B) into elements [o, o+8) of a v16bf.
__device__ inline void load8(v16bf& v, int o, const unsigned short* p) {
    US8 t = *(const US8*)p;
    unsigned short* d = reinterpret_cast<unsigned short*>(&v);
#pragma unroll
    for (int i = 0; i < 8; ++i) d[o + i] = t.s[i];
}

__device__ inline v16bf load16(const unsigned short* p) {
    v16bf v;
    load8(v, 0, p);
    load8(v, 8, p + 8);
    return v;
}

__device__ inline v8f wmma_bf16(v16bf a, v16bf b, v8f c) {
    return __builtin_amdgcn_wmma_f32_16x16x32_bf16(
        false, a, false, b, (short)0, c, false, false);
}

// Row reductions across a 16-lane half (wave32; xor masks < 16 stay in-half).
__device__ inline float rmax16(float v) {
    v = fmaxf(v, __shfl_xor(v, 1, 32));
    v = fmaxf(v, __shfl_xor(v, 2, 32));
    v = fmaxf(v, __shfl_xor(v, 4, 32));
    v = fmaxf(v, __shfl_xor(v, 8, 32));
    return v;
}
__device__ inline float rsum16(float v) {
    v += __shfl_xor(v, 1, 32);
    v += __shfl_xor(v, 2, 32);
    v += __shfl_xor(v, 4, 32);
    v += __shfl_xor(v, 8, 32);
    return v;
}

// ---------------------------------------------------------------------------
// Prep: fp32 -> bf16 convert / transpose-convert
// ---------------------------------------------------------------------------
__global__ void cvt_bf16_kernel(const float* __restrict__ in,
                                ushort_t* __restrict__ out, int n) {
    for (int i = blockIdx.x * blockDim.x + threadIdx.x; i < n;
         i += gridDim.x * blockDim.x)
        out[i] = f2bf(in[i]);
}

// out[n*rows + k] = in[k*cols + n]  (weights -> N-major bf16)
__global__ void transpose_w_kernel(const float* __restrict__ in,
                                   ushort_t* __restrict__ out,
                                   int rows, int cols) {
    int total = rows * cols;
    for (int i = blockIdx.x * blockDim.x + threadIdx.x; i < total;
         i += gridDim.x * blockDim.x) {
        int n = i / rows, k = i % rows;
        out[i] = f2bf(in[k * cols + n]);
    }
}

// ---------------------------------------------------------------------------
// QKV projection: block = 256x64 output tile, 8 waves each own 32x64.
// B-tile (64 cols x 32 k, 4KB) shared via LDS, double-buffered async.
// ---------------------------------------------------------------------------
__global__ void qkv_gemm_kernel(const ushort_t* __restrict__ xb,
                                const ushort_t* __restrict__ wT,
                                const float* __restrict__ bias,
                                ushort_t* __restrict__ Qb,
                                ushort_t* __restrict__ Kb,
                                ushort_t* __restrict__ Vt) {
    __shared__ alignas(16) ushort_t ldsB[2][64 * 32];
    const int NSTRIPS = QKVN / 64;  // 36
    int mb = blockIdx.x / NSTRIPS;
    int ns = blockIdx.x % NSTRIPS;
    int tid = threadIdx.x;
    int w = tid >> 5, lane = tid & 31, lo = lane & 15, hi = lane >> 4;
    int nbase = ns * 64;

    // cooperative B-tile source: thread -> (col, k) 8-element chunk
    int colL = (tid * 8) >> 5, kk = (tid * 8) & 31;
    const ushort_t* bsrc = wT + (size_t)(nbase + colL) * Ee + kk;

    int rbase = mb * 256 + w * 32;
    const ushort_t* arow0 = xb + (size_t)(rbase + lo) * Ee;
    const ushort_t* arow1 = xb + (size_t)(rbase + 16 + lo) * Ee;

    v8f acc[8] = {};
    cp16_to_lds(bsrc, &ldsB[0][tid * 8]);   // prologue: B tile, k-step 0

    // prologue: A fragments, k-step 0
    v16bf a0_cur = load16(arow0 + hi * 8);   // note: chunks handled below
    v16bf a1_cur;
    load8(a0_cur, 0, arow0 + hi * 16);
    load8(a0_cur, 8, arow0 + hi * 16 + 8);
    load8(a1_cur, 0, arow1 + hi * 16);
    load8(a1_cur, 8, arow1 + hi * 16 + 8);

#pragma unroll 1
    for (int kt = 0; kt < Ee / 32; ++kt) {
        int buf = kt & 1;
        __syncthreads();                     // prev consumers of buf^1 done
        if (kt + 1 < Ee / 32) {
            cp16_to_lds(bsrc + (kt + 1) * 32, &ldsB[buf ^ 1][tid * 8]);
            wait_async<1>();
        } else {
            wait_async<0>();
        }
        __syncthreads();                     // buf visible to all waves

        // prefetch next A fragments (global) behind this step's WMMAs
        v16bf a0_nxt = a0_cur, a1_nxt = a1_cur;
        if (kt + 1 < Ee / 32) {
            int k0n = (kt + 1) * 32;
            load8(a0_nxt, 0, arow0 + k0n + hi * 16);
            load8(a0_nxt, 8, arow0 + k0n + hi * 16 + 8);
            load8(a1_nxt, 0, arow1 + k0n + hi * 16);
            load8(a1_nxt, 8, arow1 + k0n + hi * 16 + 8);
        }

        v16bf b0 = load16(&ldsB[buf][(0 * 16 + lo) * 32 + hi * 16]);
        v16bf b1 = load16(&ldsB[buf][(1 * 16 + lo) * 32 + hi * 16]);
        v16bf b2 = load16(&ldsB[buf][(2 * 16 + lo) * 32 + hi * 16]);
        v16bf b3 = load16(&ldsB[buf][(3 * 16 + lo) * 32 + hi * 16]);

        acc[0] = wmma_bf16(a0_cur, b0, acc[0]);
        acc[4] = wmma_bf16(a1_cur, b0, acc[4]);
        acc[1] = wmma_bf16(a0_cur, b1, acc[1]);
        acc[5] = wmma_bf16(a1_cur, b1, acc[5]);
        acc[2] = wmma_bf16(a0_cur, b2, acc[2]);
        acc[6] = wmma_bf16(a1_cur, b2, acc[6]);
        acc[3] = wmma_bf16(a0_cur, b3, acc[3]);
        acc[7] = wmma_bf16(a1_cur, b3, acc[7]);

        a0_cur = a0_nxt;
        a1_cur = a1_nxt;
    }

#pragma unroll
    for (int g = 0; g < 2; ++g) {
#pragma unroll
        for (int t = 0; t < 4; ++t) {
            int col = nbase + t * 16 + lo;
            float bv = bias[col];
            int which = col / Ee;            // 0=Q 1=K 2=V
            int rem = col % Ee;
            int h = rem / Dd, d = rem % Dd;
#pragma unroll
            for (int i = 0; i < 8; ++i) {
                int row = rbase + g * 16 + i + 8 * hi;
                int bidx = row >> 10, nq = row & (Nn - 1);
                int bh = bidx * Hh + h;
                float v = acc[g * 4 + t][i] + bv;
                if (which == 0) {
                    Qb[((size_t)bh * Nn + nq) * Dd + d] = f2bf(v * 0.125f);
                } else if (which == 1) {
                    Kb[((size_t)bh * Nn + nq) * Dd + d] = f2bf(v);
                } else {
                    Vt[((size_t)bh * Dd + d) * Nn + nq] = f2bf(v);
                }
            }
        }
    }
}

// ---------------------------------------------------------------------------
// Flash attention: block = 8 query tiles of one (b,h); K/V tiles shared in
// LDS (double-buffered async), online softmax, P transposed C->A via LDS.
// ---------------------------------------------------------------------------
__global__ void attn_kernel(const ushort_t* __restrict__ Qb,
                            const ushort_t* __restrict__ Kb,
                            const ushort_t* __restrict__ Vt,
                            ushort_t* __restrict__ Aout) {
    __shared__ alignas(16) ushort_t ldsK[2][32 * 64];   // [key_local][d]
    __shared__ alignas(16) ushort_t ldsV[2][64 * 32];   // [d][key_local]
    __shared__ alignas(16) ushort_t ldsP[8][16 * 32];   // per-wave P staging

    int tid = threadIdx.x;
    int w = tid >> 5, lane = tid & 31, lo = lane & 15, hi = lane >> 4;
    int bh = blockIdx.x >> 3;
    int qt = (blockIdx.x & 7) * 8 + w;

    const ushort_t* qp = Qb + (size_t)bh * Nn * Dd;
    const ushort_t* kp = Kb + (size_t)bh * Nn * Dd;
    const ushort_t* vp = Vt + (size_t)bh * Dd * Nn;
    ushort_t* op = Aout + (size_t)bh * Nn * Dd;

    // cooperative tile sources
    int kl = (tid * 8) >> 6, kd = (tid * 8) & 63;   // K: (key_local, d_off)
    int vd = (tid * 8) >> 5, vk = (tid * 8) & 31;   // V: (d_row, key_off)

    // Q fragments (d = 0..31, 32..63), rows qt*16..qt*16+15
    int qrow = qt * 16 + lo;
    v16bf a0, a1;
    load8(a0, 0, qp + (size_t)qrow * Dd + hi * 16);
    load8(a0, 8, qp + (size_t)qrow * Dd + hi * 16 + 8);
    load8(a1, 0, qp + (size_t)qrow * Dd + 32 + hi * 16);
    load8(a1, 8, qp + (size_t)qrow * Dd + 32 + hi * 16 + 8);

    float mrow[8], lrow[8];
    v8f o[4] = {};
#pragma unroll
    for (int i = 0; i < 8; ++i) { mrow[i] = -1e30f; lrow[i] = 0.f; }

    // prologue: stage tiles for j = 0
    cp16_to_lds(kp + (size_t)kl * Dd + kd, &ldsK[0][tid * 8]);
    cp16_to_lds(vp + (size_t)vd * Nn + vk, &ldsV[0][tid * 8]);

#pragma unroll 1
    for (int idx = 0; idx < Nn / 32; ++idx) {
        int j = idx * 32;
        int buf = idx & 1;
        __syncthreads();                       // prev consumers of buf^1 done
        if (idx + 1 < Nn / 32) {
            cp16_to_lds(kp + (size_t)(j + 32 + kl) * Dd + kd, &ldsK[buf ^ 1][tid * 8]);
            cp16_to_lds(vp + (size_t)vd * Nn + j + 32 + vk, &ldsV[buf ^ 1][tid * 8]);
            wait_async<2>();
        } else {
            wait_async<0>();
        }
        __syncthreads();                       // tiles visible to all waves

        // ---- energy: batch 4 K fragments, then 4 WMMAs (2 indep chains) ----
        v16bf bk00 = load16(&ldsK[buf][lo * 64 + hi * 16]);
        v16bf bk01 = load16(&ldsK[buf][lo * 64 + 32 + hi * 16]);
        v16bf bk10 = load16(&ldsK[buf][(16 + lo) * 64 + hi * 16]);
        v16bf bk11 = load16(&ldsK[buf][(16 + lo) * 64 + 32 + hi * 16]);
        v8f e0 = {}, e1 = {};
        e0 = wmma_bf16(a0, bk00, e0);
        e1 = wmma_bf16(a0, bk10, e1);
        e0 = wmma_bf16(a1, bk01, e0);
        e1 = wmma_bf16(a1, bk11, e1);

        // issue V fragment loads now: latency hides behind softmax VALU
        v16bf bv0 = load16(&ldsV[buf][(0 * 16 + lo) * 32 + hi * 16]);
        v16bf bv1 = load16(&ldsV[buf][(1 * 16 + lo) * 32 + hi * 16]);
        v16bf bv2 = load16(&ldsV[buf][(2 * 16 + lo) * 32 + hi * 16]);
        v16bf bv3 = load16(&ldsV[buf][(3 * 16 + lo) * 32 + hi * 16]);

        // ---- online softmax (row = i + 8*hi) ----
#pragma unroll
        for (int i = 0; i < 8; ++i) {
            float rm = rmax16(fmaxf(e0[i], e1[i]));
            float nm = fmaxf(mrow[i], rm);
            float sc = __expf(mrow[i] - nm);
            float p0 = __expf(e0[i] - nm);
            float p1 = __expf(e1[i] - nm);
            lrow[i] = lrow[i] * sc + rsum16(p0 + p1);
            mrow[i] = nm;
            o[0][i] *= sc; o[1][i] *= sc; o[2][i] *= sc; o[3][i] *= sc;
            int rr = i + 8 * hi;
            ldsP[w][rr * 32 + lo]      = f2bf(p0);
            ldsP[w][rr * 32 + 16 + lo] = f2bf(p1);
        }
        asm volatile("s_wait_dscnt 0x0" ::: "memory");   // intra-wave LDS RAW

        // P as A-fragment 16x32 (rows = queries, k = keys)
        v16bf pa;
        load8(pa, 0, &ldsP[w][lo * 32 + hi * 16]);
        load8(pa, 8, &ldsP[w][lo * 32 + hi * 16 + 8]);

        // ---- O += P @ V ----
        o[0] = wmma_bf16(pa, bv0, o[0]);
        o[1] = wmma_bf16(pa, bv1, o[1]);
        o[2] = wmma_bf16(pa, bv2, o[2]);
        o[3] = wmma_bf16(pa, bv3, o[3]);
    }

    // normalize + store (flat layout == reference's reshape quirk)
#pragma unroll
    for (int t = 0; t < 4; ++t) {
#pragma unroll
        for (int i = 0; i < 8; ++i) {
            int row = qt * 16 + i + 8 * hi;
            int d = t * 16 + lo;
            op[(size_t)row * Dd + d] = f2bf(o[t][i] / lrow[i]);
        }
    }
}

// ---------------------------------------------------------------------------
// Output projection: block = 256x64 tile, 8 waves each own 32x64, fp32 out
// ---------------------------------------------------------------------------
__global__ void out_gemm_kernel(const ushort_t* __restrict__ ab,
                                const ushort_t* __restrict__ wT,
                                const float* __restrict__ bias,
                                float* __restrict__ out) {
    __shared__ alignas(16) ushort_t ldsB[2][64 * 32];
    const int NSTRIPS = Ee / 64;  // 12
    int mb = blockIdx.x / NSTRIPS;
    int ns = blockIdx.x % NSTRIPS;
    int tid = threadIdx.x;
    int w = tid >> 5, lane = tid & 31, lo = lane & 15, hi = lane >> 4;
    int nbase = ns * 64;

    int colL = (tid * 8) >> 5, kk = (tid * 8) & 31;
    const ushort_t* bsrc = wT + (size_t)(nbase + colL) * Ee + kk;

    int rbase = mb * 256 + w * 32;
    const ushort_t* arow0 = ab + (size_t)(rbase + lo) * Ee;
    const ushort_t* arow1 = ab + (size_t)(rbase + 16 + lo) * Ee;

    v8f acc[8] = {};
    cp16_to_lds(bsrc, &ldsB[0][tid * 8]);

    v16bf a0_cur, a1_cur;
    load8(a0_cur, 0, arow0 + hi * 16);
    load8(a0_cur, 8, arow0 + hi * 16 + 8);
    load8(a1_cur, 0, arow1 + hi * 16);
    load8(a1_cur, 8, arow1 + hi * 16 + 8);

#pragma unroll 1
    for (int kt = 0; kt < Ee / 32; ++kt) {
        int buf = kt & 1;
        __syncthreads();
        if (kt + 1 < Ee / 32) {
            cp16_to_lds(bsrc + (kt + 1) * 32, &ldsB[buf ^ 1][tid * 8]);
            wait_async<1>();
        } else {
            wait_async<0>();
        }
        __syncthreads();

        v16bf a0_nxt = a0_cur, a1_nxt = a1_cur;
        if (kt + 1 < Ee / 32) {
            int k0n = (kt + 1) * 32;
            load8(a0_nxt, 0, arow0 + k0n + hi * 16);
            load8(a0_nxt, 8, arow0 + k0n + hi * 16 + 8);
            load8(a1_nxt, 0, arow1 + k0n + hi * 16);
            load8(a1_nxt, 8, arow1 + k0n + hi * 16 + 8);
        }

        v16bf b0 = load16(&ldsB[buf][(0 * 16 + lo) * 32 + hi * 16]);
        v16bf b1 = load16(&ldsB[buf][(1 * 16 + lo) * 32 + hi * 16]);
        v16bf b2 = load16(&ldsB[buf][(2 * 16 + lo) * 32 + hi * 16]);
        v16bf b3 = load16(&ldsB[buf][(3 * 16 + lo) * 32 + hi * 16]);

        acc[0] = wmma_bf16(a0_cur, b0, acc[0]);
        acc[4] = wmma_bf16(a1_cur, b0, acc[4]);
        acc[1] = wmma_bf16(a0_cur, b1, acc[1]);
        acc[5] = wmma_bf16(a1_cur, b1, acc[5]);
        acc[2] = wmma_bf16(a0_cur, b2, acc[2]);
        acc[6] = wmma_bf16(a1_cur, b2, acc[6]);
        acc[3] = wmma_bf16(a0_cur, b3, acc[3]);
        acc[7] = wmma_bf16(a1_cur, b3, acc[7]);

        a0_cur = a0_nxt;
        a1_cur = a1_nxt;
    }

#pragma unroll
    for (int g = 0; g < 2; ++g) {
#pragma unroll
        for (int t = 0; t < 4; ++t) {
            int col = nbase + t * 16 + lo;
            float bv = bias[col];
#pragma unroll
            for (int i = 0; i < 8; ++i) {
                int row = rbase + g * 16 + i + 8 * hi;
                out[(size_t)row * Ee + col] = acc[g * 4 + t][i] + bv;
            }
        }
    }
}

// ---------------------------------------------------------------------------
extern "C" void kernel_launch(void* const* d_in, const int* in_sizes, int n_in,
                              void* d_out, int out_size, void* d_ws, size_t ws_size,
                              hipStream_t stream) {
    const float* x    = (const float*)d_in[0];
    const float* Wqkv = (const float*)d_in[1];
    const float* bqkv = (const float*)d_in[2];
    const float* Wout = (const float*)d_in[3];
    const float* bout = (const float*)d_in[4];
    float* out = (float*)d_out;

    const size_t SZ = (size_t)Bb * Nn * Ee;     // 12,582,912 elements
    ushort_t* ws  = (ushort_t*)d_ws;
    ushort_t* xb  = ws;                          // x bf16 [16384,768]
    ushort_t* Qb  = ws + 1 * SZ;                 // [B,H,N,D]
    ushort_t* Kb  = ws + 2 * SZ;                 // [B,H,N,D]
    ushort_t* Vt  = ws + 3 * SZ;                 // [B,H,D,N]
    ushort_t* Ab  = ws + 4 * SZ;                 // attn out bf16 [16384,768]
    ushort_t* WqT = ws + 5 * SZ;                 // [2304,768]
    ushort_t* WoT = WqT + (size_t)QKVN * Ee;     // [768,768]

    cvt_bf16_kernel<<<4096, 256, 0, stream>>>(x, xb, (int)SZ);
    transpose_w_kernel<<<2048, 256, 0, stream>>>(Wqkv, WqT, Ee, QKVN);
    transpose_w_kernel<<<1024, 256, 0, stream>>>(Wout, WoT, Ee, Ee);

    // 64 M-blocks (256 rows) x 36 N-strips (64 cols)
    qkv_gemm_kernel<<<64 * 36, 256, 0, stream>>>(xb, WqT, bqkv, Qb, Kb, Vt);

    // 16*12 heads x 8 blocks of 8 query tiles
    attn_kernel<<<Bb * Hh * 8, 256, 0, stream>>>(Qb, Kb, Vt, Ab);

    // 64 M-blocks x 12 N-strips
    out_gemm_kernel<<<64 * 12, 256, 0, stream>>>(Ab, WoT, bout, out);
}
